// DeformConv2D_27230092657029
// MI455X (gfx1250) — matrix operand order repaired and checked by reading, not compiled, not run on hardware
//
#include <hip/hip_runtime.h>

#define KS    3
#define NPTS  9            // KS*KS sampling points
#define B_    8
#define C_    64
#define H_    96
#define W_    96
#define OUTC_ 64
#define HW_   (H_ * W_)            // 9216
#define KRED  (C_ * NPTS)          // 576 (GEMM reduction)
#define MROWS (B_ * HW_)           // 73728 (GEMM rows / pixels)
#define KTILES (KRED / 32)         // 18
#define NTILES (OUTC_ / 16)        // 4
#define BELEMS (NTILES * KTILES * 32 * 16)   // 36864 f16 = 73728 bytes

typedef __attribute__((ext_vector_type(16))) _Float16 v16h;
typedef __attribute__((ext_vector_type(8)))  _Float16 v8h;
typedef __attribute__((ext_vector_type(8)))  float    v8f;
typedef __attribute__((ext_vector_type(4)))  unsigned int u32x4;
typedef __attribute__((ext_vector_type(4)))  int i32x4;
typedef __attribute__((ext_vector_type(8)))  int i32x8;

// ---------------------------------------------------------------------------
// Kernel 0: pack w_ker (f32, [OUTC][C][3][3]) into the wave32 WMMA B layout,
// f16. Flat index = ((j*KTILES + kt)*32 + lane)*16 + e, where the lane holds
// column oc = j*16 + (lane&15) and K = kt*32 + (lane>>4)*8 + (e&7) + (e>>3)*16.
// ---------------------------------------------------------------------------
__global__ __launch_bounds__(256) void pack_b_kernel(
    const float* __restrict__ w_ker, _Float16* __restrict__ packedB) {
  int tid = blockIdx.x * 256 + threadIdx.x;
  if (tid >= BELEMS) return;
  int e  = tid & 15;
  int l  = (tid >> 4) & 31;
  int kt = (tid >> 9) % KTILES;
  int j  = tid / (512 * KTILES);
  int k  = kt * 32 + (l >> 4) * 8 + (e & 7) + (e >> 3) * 16;      // 0..575
  int oc = j * 16 + (l & 15);
  packedB[tid] = (_Float16)w_ker[(size_t)oc * KRED + k];
}

// ---------------------------------------------------------------------------
// Kernel 1: offset conv (18ch, 3x3 pad1) + bilinear sampling -> f16 im2col A.
// One thread per pixel. A[p][c*9+n], row stride KRED=576 f16 (1152B).
// ---------------------------------------------------------------------------
__global__ __launch_bounds__(256) void deform_sample_kernel(
    const float* __restrict__ x, const float* __restrict__ w_off,
    const float* __restrict__ b_off, _Float16* __restrict__ A) {
  __shared__ float s_woff[2 * NPTS * C_ * KS * KS];   // 18*64*9 = 10368 floats
  for (int i = threadIdx.x; i < 2 * NPTS * C_ * KS * KS; i += 256)
    s_woff[i] = w_off[i];
  __syncthreads();

  int p   = blockIdx.x * 256 + threadIdx.x;           // grid sized exactly
  int b   = p / HW_;
  int rem = p - b * HW_;
  int h   = rem / W_;
  int w   = rem - h * W_;

  // ---- offset field: 18-channel 3x3 conv at (b,h,w) ----
  float acc[2 * NPTS];
  #pragma unroll
  for (int o = 0; o < 2 * NPTS; ++o) acc[o] = b_off[o];

  const float* xb = x + (size_t)b * C_ * HW_;
  #pragma unroll 2
  for (int c = 0; c < C_; ++c) {
    const float* xc = xb + c * HW_;
    #pragma unroll
    for (int kr = 0; kr < 3; ++kr) {
      int hh = h + kr - 1;
      if (hh < 0 || hh >= H_) continue;
      #pragma unroll
      for (int kc = 0; kc < 3; ++kc) {
        int ww = w + kc - 1;
        if (ww < 0 || ww >= W_) continue;
        float xv = xc[hh * W_ + ww];
        #pragma unroll
        for (int o = 0; o < 2 * NPTS; ++o)
          acc[o] += xv * s_woff[((o * C_ + c) * 3 + kr) * 3 + kc];
      }
    }
  }

  // ---- per-point bilinear corner indices & weights (shared across c) ----
  int   i00[NPTS], i10[NPTS], i01[NPTS], i11[NPTS];
  float c00[NPTS], c10[NPTS], c01[NPTS], c11[NPTS];
  #pragma unroll
  for (int n = 0; n < NPTS; ++n) {
    float px = (float)h + (float)(n / 3 - 1) + acc[n];        // row coord
    float py = (float)w + (float)(n % 3 - 1) + acc[n + NPTS]; // col coord
    float x0f = floorf(px), y0f = floorf(py);
    float lx = px - x0f, ly = py - y0f;
    int x0 = (int)x0f, y0 = (int)y0f;
    int x1 = x0 + 1,  y1 = y0 + 1;
    float vx0 = (x0 >= 0 && x0 < H_) ? 1.f : 0.f;
    float vx1 = (x1 >= 0 && x1 < H_) ? 1.f : 0.f;
    float vy0 = (y0 >= 0 && y0 < W_) ? 1.f : 0.f;
    float vy1 = (y1 >= 0 && y1 < W_) ? 1.f : 0.f;
    int cx0 = min(max(x0, 0), H_ - 1), cx1 = min(max(x1, 0), H_ - 1);
    int cy0 = min(max(y0, 0), W_ - 1), cy1 = min(max(y1, 0), W_ - 1);
    i00[n] = cx0 * W_ + cy0;  c00[n] = (1.f - lx) * (1.f - ly) * vx0 * vy0;
    i10[n] = cx1 * W_ + cy0;  c10[n] = lx * (1.f - ly) * vx1 * vy0;
    i01[n] = cx0 * W_ + cy1;  c01[n] = (1.f - lx) * ly * vx0 * vy1;
    i11[n] = cx1 * W_ + cy1;  c11[n] = lx * ly * vx1 * vy1;
  }

  // ---- gather + weight -> f16 im2col row ----
  _Float16* Arow = A + (size_t)p * KRED;
  #pragma unroll 2
  for (int c = 0; c < C_; ++c) {
    const float* xc = xb + c * HW_;
    #pragma unroll
    for (int n = 0; n < NPTS; ++n) {
      float val = c00[n] * xc[i00[n]] + c10[n] * xc[i10[n]] +
                  c01[n] * xc[i01[n]] + c11[n] * xc[i11[n]];
      Arow[c * NPTS + n] = (_Float16)val;
    }
  }
}

// ---------------------------------------------------------------------------
// Kernel 2: GEMM via v_wmma_f32_16x16x32_f16.
// Block = 256 threads = 8 waves. Wave i owns M-tile (blockIdx.x*8+i) and
// computes all 4 oc-tiles, reusing its A registers 4x. The full packed B
// (72 KB) is staged into LDS once per block via the Tensor Data Mover and
// read through ds_load for the WMMA B operand. A is read from HBM exactly
// once per row.
// ---------------------------------------------------------------------------
__global__ __launch_bounds__(256) void gemm_wmma_kernel(
    const _Float16* __restrict__ A, const _Float16* __restrict__ packedB,
    const float* __restrict__ b_ker, float* __restrict__ out) {
  __shared__ _Float16 sB[BELEMS];   // 73728 bytes

#if __has_builtin(__builtin_amdgcn_tensor_load_to_lds)
  if (threadIdx.x < 32) {           // wave-uniform: only wave 0 issues the TDM
    unsigned lds_off = (unsigned)(uintptr_t)(void*)&sB[0];  // low 32 bits = LDS offset
    unsigned long long ga = (unsigned long long)(uintptr_t)packedB;
    // D# group0: count=1, lds_addr, global_addr[56:0], type=2 ("image")
    u32x4 g0 = { 1u,
                 lds_off,
                 (unsigned)(ga & 0xffffffffu),
                 (unsigned)((ga >> 32) & 0x01ffffffu) | (2u << 30) };
    // D# group1: data_size=4B (code 2); 1 row of 18432 dwords (73728 B tile)
    i32x8 g1 = { (int)(2u << 16),        // [17:16] data_size
                 (int)(0x4800u << 16),   // [63:48] tensor_dim0[15:0] = 18432
                 (int)(1u << 16),        // [95:80] tensor_dim1[15:0] = 1
                 (int)(0x4800u << 16),   // [127:112] tile_dim0 = 18432
                 1,                      // [143:128] tile_dim1 = 1
                 18432,                  // [191:160] tensor_dim0_stride lo
                 (int)(0x4800u << 16),   // [223:208] tensor_dim1_stride[15:0]
                 0 };
    i32x4 gz = { 0, 0, 0, 0 };
#if defined(__clang_major__) && (__clang_major__ >= 23)
    i32x8 gz8 = { 0, 0, 0, 0, 0, 0, 0, 0 };
    __builtin_amdgcn_tensor_load_to_lds(g0, g1, gz, gz, gz8, 0);
#else
    __builtin_amdgcn_tensor_load_to_lds(g0, g1, gz, gz, 0);
#endif
    __builtin_amdgcn_s_wait_tensorcnt(0);
  }
#else
  for (int i = threadIdx.x; i < BELEMS / 8; i += 256)
    ((v8h*)sB)[i] = ((const v8h*)packedB)[i];
#endif
  __syncthreads();

  int lane = threadIdx.x & 31;
  int wv   = threadIdx.x >> 5;      // wave -> M-tile within block
  int hf   = lane >> 4;             // half-wave select
  int lrow = lane & 15;

  int mrow = (blockIdx.x * 8 + wv) * 16 + lrow;
  const _Float16* Arow = A + (size_t)mrow * KRED;

  v8f acc[NTILES] = {};
  #pragma unroll
  for (int kt = 0; kt < KTILES; ++kt) {
    // 16-bit A layout: lane holds K = base..base+7 and base+16..base+23,
    // base = kt*32 + 8*hf (16B-aligned chunks).
    int kbase = kt * 32 + hf * 8;
    v8h alo = *(const v8h*)(Arow + kbase);
    v8h ahi = *(const v8h*)(Arow + kbase + 16);
    v16h a = __builtin_shufflevector(alo, ahi,
        0, 1, 2, 3, 4, 5, 6, 7, 8, 9, 10, 11, 12, 13, 14, 15);
    #pragma unroll
    for (int j = 0; j < NTILES; ++j) {
      v16h bv = *(const v16h*)(&sB[(((size_t)j * KTILES + kt) * 32 + lane) * 16]);
      acc[j] = __builtin_amdgcn_wmma_f32_16x16x32_f16(
          false, a, false, bv, (short)0, acc[j], false, false);
    }
  }

  #pragma unroll
  for (int j = 0; j < NTILES; ++j) {
    int oc = j * 16 + lrow;
    float bias = b_ker[oc];
    #pragma unroll
    for (int v = 0; v < 8; ++v) {
      int p   = (blockIdx.x * 8 + wv) * 16 + v + 8 * hf;  // D: M=v+8*hf, N=lrow
      int b   = p / HW_;
      int rem = p - b * HW_;
      out[((size_t)b * OUTC_ + oc) * HW_ + rem] = acc[j][v] + bias;
    }
  }
}

// ---------------------------------------------------------------------------
extern "C" void kernel_launch(void* const* d_in, const int* in_sizes, int n_in,
                              void* d_out, int out_size, void* d_ws, size_t ws_size,
                              hipStream_t stream) {
  const float* x     = (const float*)d_in[0];
  const float* w_off = (const float*)d_in[1];
  const float* b_off = (const float*)d_in[2];
  const float* w_ker = (const float*)d_in[3];
  const float* b_ker = (const float*)d_in[4];
  float* out = (float*)d_out;

  // ws layout: [packedB: 36864 f16 = 73728 B (256B-aligned)] [A: 73728*576 f16]
  _Float16* packedB = (_Float16*)d_ws;
  _Float16* A = (_Float16*)((char*)d_ws + (size_t)BELEMS * sizeof(_Float16));

  pack_b_kernel<<<(BELEMS + 255) / 256, 256, 0, stream>>>(w_ker, packedB);
  deform_sample_kernel<<<MROWS / 256, 256, 0, stream>>>(x, w_off, b_off, A);
  gemm_wmma_kernel<<<MROWS / 128, 256, 0, stream>>>(A, packedB, b_ker, out);
}
